// AttentionBlock_62397284876438
// MI455X (gfx1250) — matrix operand order repaired and checked by reading, not compiled, not run on hardware
//
#include <hip/hip_runtime.h>
#include <hip/hip_bf16.h>

// ---------------- problem constants ----------------
#define BATCH 32
#define CCH   256        // channels
#define NSEQ  1024       // H*W
#define GRP   8
#define CG    (CCH / GRP)        // 32 channels per group
#define O3    (3 * CCH)          // 768 qkv output rows
#define EPSV  1e-5f

typedef __attribute__((ext_vector_type(16))) __bf16 bf16x16;
typedef __attribute__((ext_vector_type(8)))  float  f32x8;

// ---------------------------------------------------------------------------
// CDNA5 async global->LDS copy (ASYNCcnt-tracked), with safe fallback.
// Builtin prototype (from hipcc diagnostic): (global int4*, lds int4*, Ii, Ii)
// ---------------------------------------------------------------------------
#if __has_builtin(__builtin_amdgcn_global_load_async_to_lds_b128)
#define HAVE_ASYNC_LDS 1
typedef int v4i_gcc __attribute__((vector_size(16)));
typedef __attribute__((address_space(1))) v4i_gcc* as1_v4i;
typedef __attribute__((address_space(3))) v4i_gcc* as3_v4i;
#endif

__device__ __forceinline__ void copy16_g2l(const __bf16* g, __bf16* l) {
#ifdef HAVE_ASYNC_LDS
    __builtin_amdgcn_global_load_async_to_lds_b128(
        (as1_v4i)(uintptr_t)g, (as3_v4i)(uintptr_t)l, 0, 0);
#else
    *reinterpret_cast<uint4*>(l) = *reinterpret_cast<const uint4*>(g);
#endif
}

template <int N>
__device__ __forceinline__ void wait_async() {
#ifdef HAVE_ASYNC_LDS
#if __has_builtin(__builtin_amdgcn_s_wait_asynccnt)
    __builtin_amdgcn_s_wait_asynccnt((unsigned short)N);
#else
    asm volatile("s_wait_asynccnt %0" ::"i"(N) : "memory");
#endif
#endif
}

// ---------------------------------------------------------------------------
// Fragment loaders (wave32 WMMA; this code's contiguous K order, same for A and B)
// A 16x32 bf16 : lane half h, row m=lane%16 ; elems 0..15 = K k0+16h+0..15
//                (k0 = 32*kk)
// B 32x16 bf16 : lane half h, col n=lane%16 ; elems 0..15 = K k0+16h+0..15
// Contiguous per-lane when the K dimension is contiguous in memory/LDS.
// ---------------------------------------------------------------------------
__device__ __forceinline__ bf16x16 frag_a_bf16(const __bf16* __restrict__ row,
                                               int kk, int half) {
    bf16x16 a;
    const __bf16* p = row + kk * 32 + half * 16;
#pragma unroll
    for (int e = 0; e < 16; ++e) a[e] = p[e];
    return a;
}

__device__ __forceinline__ bf16x16 frag_b_bf16(const __bf16* __restrict__ col,
                                               int kk, int half) {
    bf16x16 b;
    const __bf16* p = col + kk * 32 + half * 16;
#pragma unroll
    for (int e = 0; e < 16; ++e) b[e] = p[e];
    return b;
}

__device__ __forceinline__ bf16x16 frag_b_lds(const __bf16* p) {
    bf16x16 b;
#pragma unroll
    for (int e = 0; e < 16; ++e) b[e] = p[e];
    return b;
}

#define WMMA_BF16(A, B, C)                                                    \
    __builtin_amdgcn_wmma_f32_16x16x32_bf16(false, (A), false, (B), (short)0, \
                                            (C), false, false)

// ---------------------------------------------------------------------------
// Kernel 0: one-shot fp32 -> bf16 weight conversion (qkv_w 768x256,
// proj_w 256x256).  Removes per-tile cvt VALU from the GEMM inner loops.
// ---------------------------------------------------------------------------
__global__ __launch_bounds__(256)
void wcvt_kernel(const float* __restrict__ qkv_w, const float* __restrict__ proj_w,
                 __bf16* __restrict__ wq, __bf16* __restrict__ wp) {
    int i = blockIdx.x * 256 + threadIdx.x;
    if (i < O3 * CCH)  wq[i] = (__bf16)qkv_w[i];
    if (i < CCH * CCH) wp[i] = (__bf16)proj_w[i];
}

// ---------------------------------------------------------------------------
// Kernel 1: GroupNorm, fp32 in -> bf16 out, transposed to (B, N, C) so the
// QKV GEMM B-operand has contiguous K(=C). One block per (b, g).
// ---------------------------------------------------------------------------
__global__ __launch_bounds__(256)
void gn_kernel(const float* __restrict__ x, const float* __restrict__ gw,
               const float* __restrict__ gb, __bf16* __restrict__ xnT) {
    const int b = blockIdx.x / GRP;
    const int g = blockIdx.x % GRP;
    const int t = threadIdx.x;
    const int GE = CG * NSEQ;                       // 32768 elems per group
    const float* xg = x + ((size_t)b * CCH + g * CG) * NSEQ;

    float s = 0.f, ss = 0.f;
    for (int i = t; i < GE; i += 256) { float v = xg[i]; s += v; ss += v * v; }

    __shared__ float rs[256], rss[256];
    rs[t] = s; rss[t] = ss; __syncthreads();
#pragma unroll
    for (int off = 128; off > 0; off >>= 1) {
        if (t < off) { rs[t] += rs[t + off]; rss[t] += rss[t + off]; }
        __syncthreads();
    }
    const float mean = rs[0] / (float)GE;
    const float var  = rss[0] / (float)GE - mean * mean;
    const float rinv = rsqrtf(var + EPSV);

    // c-major iteration -> coalesced transposed stores; re-reads hit L2
    for (int i = t; i < GE; i += 256) {
        int cl = i & (CG - 1);
        int n  = i >> 5;                             // CG == 32
        int c  = g * CG + cl;
        float v = xg[(size_t)cl * NSEQ + n];
        float y = (v - mean) * rinv * gw[c] + gb[c];
        xnT[((size_t)b * NSEQ + n) * CCH + c] = (__bf16)y;
    }
}

// ---------------------------------------------------------------------------
// Kernel 2: QKV GEMM.  qkv[b,o,n] = sum_c W[o,c]*xn[b,c,n] + bias[o]
// One wave per 32x32 output tile (2x2 WMMA tiles, A/B fragment reuse),
// K = 256 in 8 chunks of 32.
// q (o<256)  -> qT (B,N,C), scaled by C^-0.5 (folded softmax scale)
// k          -> kT (B,N,C);   v -> v (B,C,N)
// ---------------------------------------------------------------------------
__global__ __launch_bounds__(128)
void qkv_kernel(const __bf16* __restrict__ xnT, const __bf16* __restrict__ W,
                const float* __restrict__ bias, __bf16* __restrict__ qT,
                __bf16* __restrict__ kT, __bf16* __restrict__ v) {
    const int wid  = threadIdx.x >> 5;
    const int lane = threadIdx.x & 31;
    const int half = lane >> 4;
    const int l16  = lane & 15;

    const int tile = blockIdx.x * 4 + wid;          // 32 * 24 * 32 waves
    const int b    = tile / (24 * 32);
    const int rem  = tile % (24 * 32);
    const int ob   = rem / 32;                      // 32-row o block
    const int nb   = rem % 32;                      // 32-col n block

    const __bf16* arow0 = W + (size_t)(ob * 32 + l16) * CCH;
    const __bf16* arow1 = arow0 + (size_t)16 * CCH;
    const __bf16* bcol0 = xnT + ((size_t)b * NSEQ + nb * 32 + l16) * CCH;
    const __bf16* bcol1 = bcol0 + (size_t)16 * CCH;

    f32x8 d00 = {}, d01 = {}, d10 = {}, d11 = {};
#pragma unroll
    for (int kk = 0; kk < 8; ++kk) {
        bf16x16 a0 = frag_a_bf16(arow0, kk, half);
        bf16x16 a1 = frag_a_bf16(arow1, kk, half);
        bf16x16 b0 = frag_b_bf16(bcol0, kk, half);
        bf16x16 b1 = frag_b_bf16(bcol1, kk, half);
        d00 = WMMA_BF16(a0, b0, d00);
        d01 = WMMA_BF16(a0, b1, d01);
        d10 = WMMA_BF16(a1, b0, d10);
        d11 = WMMA_BF16(a1, b1, d11);
    }

    const int obase = ob * 32;                      // region-uniform per wave
#pragma unroll
    for (int ti = 0; ti < 2; ++ti) {
#pragma unroll
        for (int tj = 0; tj < 2; ++tj) {
            const f32x8& d = ti ? (tj ? d11 : d10) : (tj ? d01 : d00);
            int n = nb * 32 + tj * 16 + l16;
#pragma unroll
            for (int r = 0; r < 8; ++r) {
                int o = obase + ti * 16 + r + half * 8;
                float y = d[r] + bias[o];
                if (obase < CCH) {
                    qT[((size_t)b * NSEQ + n) * CCH + o] = (__bf16)(y * 0.0625f);
                } else if (obase < 2 * CCH) {
                    kT[((size_t)b * NSEQ + n) * CCH + (o - CCH)] = (__bf16)y;
                } else {
                    v[((size_t)b * CCH + (o - 2 * CCH)) * NSEQ + n] = (__bf16)y;
                }
            }
        }
    }
}

// ---------------------------------------------------------------------------
// Kernel 3: fused flash attention. One workgroup (4 waves) per
// (batch, 16-query tile). k/v chunks (64 keys) are double-buffered in LDS
// via CDNA5 async global->LDS loads (ASYNCcnt) so chunk i+1 streams in
// while chunk i feeds the WMMAs. Each wave: 16-key slice of the score GEMM,
// 64-channel slice of the output GEMM. Online softmax state replicated
// across the 16-lane half owning each accumulator row.
// ---------------------------------------------------------------------------
#define KV_UNITS 16   // 128 threads * 16B * 16 = 32KB per tile

__global__ __launch_bounds__(128)
void flash_kernel(const __bf16* __restrict__ qT, const __bf16* __restrict__ kT,
                  const __bf16* __restrict__ v, __bf16* __restrict__ aT) {
    const int wid  = threadIdx.x >> 5;
    const int lane = threadIdx.x & 31;
    const int half = lane >> 4;
    const int l16  = lane & 15;

    const int b  = blockIdx.x >> 6;                  // 64 query tiles / batch
    const int qt = blockIdx.x & 63;
    const int i0 = qt * 16;

    __shared__ __align__(16) __bf16 sK[2][64 * CCH / 4];   // 64 keys x 256c
    __shared__ __align__(16) __bf16 sV[2][CCH * 64 / 4];   // 256c x 64 keys
    __shared__ __bf16 sP[16][64];                          // prob tile
    __shared__ float  sMax[4][16], sSum[4][16];

    const __bf16* kbase = kT + (size_t)b * NSEQ * CCH;
    const __bf16* vbase = v + (size_t)b * CCH * NSEQ;

    auto issue_chunk = [&](int jc, int buf) {
        // k chunk: rows j = jc*64 .. +63, contiguous 32KB
        const __bf16* gk = kbase + (size_t)jc * 64 * CCH;
        __bf16* lk = &sK[buf][0];
#pragma unroll
        for (int u = 0; u < KV_UNITS; ++u) {
            int i = u * 128 + (int)threadIdx.x;      // 16B units
            copy16_g2l(gk + i * 8, lk + i * 8);
        }
        // v chunk: 256 rows of 128B, stride NSEQ*2 bytes
        const __bf16* gv = vbase + jc * 64;
        __bf16* lv = &sV[buf][0];
#pragma unroll
        for (int u = 0; u < KV_UNITS; ++u) {
            int i  = u * 128 + (int)threadIdx.x;
            int c  = i >> 3;
            int uo = i & 7;
            copy16_g2l(gv + (size_t)c * NSEQ + uo * 8, lv + i * 8);
        }
    };

    // Preload q A-fragments for all of K = C = 256 (8 chunks)
    const __bf16* qrow = qT + ((size_t)b * NSEQ + i0 + l16) * CCH;
    bf16x16 aQ[8];
#pragma unroll
    for (int kk = 0; kk < 8; ++kk) aQ[kk] = frag_a_bf16(qrow, kk, half);

    float mrow[8], lrow[8];
    f32x8 acc[4];
#pragma unroll
    for (int r = 0; r < 8; ++r) { mrow[r] = -__builtin_inff(); lrow[r] = 0.f; }
#pragma unroll
    for (int t = 0; t < 4; ++t) acc[t] = (f32x8){};

    issue_chunk(0, 0);

    for (int it = 0; it < NSEQ / 64; ++it) {
        const int buf = it & 1;
        if (it + 1 < NSEQ / 64) {
            issue_chunk(it + 1, 1 - buf);            // prefetch next chunk
            wait_async<2 * KV_UNITS>();              // chunk `it` complete
        } else {
            wait_async<0>();
        }
        __syncthreads();                             // LDS tiles visible

        // ---- scores: S(16q x 16j) for this wave's key slice, K = 256 ----
        f32x8 s = {};
#pragma unroll
        for (int kk = 0; kk < 8; ++kk) {
            bf16x16 bk = frag_b_lds(&sK[buf][(wid * 16 + l16) * CCH
                                            + kk * 32 + half * 16]);
            s = WMMA_BF16(aQ[kk], bk, s);
        }
        // ---- per-row max within the 16-lane half, then across waves ----
        float tmax[8];
#pragma unroll
        for (int r = 0; r < 8; ++r) {
            float m = s[r];
            m = fmaxf(m, __shfl_xor(m, 1));
            m = fmaxf(m, __shfl_xor(m, 2));
            m = fmaxf(m, __shfl_xor(m, 4));
            m = fmaxf(m, __shfl_xor(m, 8));
            tmax[r] = m;
        }
        if (l16 == 0) {
#pragma unroll
            for (int r = 0; r < 8; ++r) sMax[wid][r + half * 8] = tmax[r];
        }
        __syncthreads();

        float mnew[8], scl[8];
#pragma unroll
        for (int r = 0; r < 8; ++r) {
            int row = r + half * 8;
            float m64 = fmaxf(fmaxf(sMax[0][row], sMax[1][row]),
                              fmaxf(sMax[2][row], sMax[3][row]));
            mnew[r] = fmaxf(mrow[r], m64);
            scl[r]  = __expf(mrow[r] - mnew[r]);
        }
        // ---- P = exp(s - mnew); stage to LDS; row sums ----
        float tsum[8];
#pragma unroll
        for (int r = 0; r < 8; ++r) {
            float p = __expf(s[r] - mnew[r]);
            sP[r + half * 8][wid * 16 + l16] = (__bf16)p;
            float q = p;
            q += __shfl_xor(q, 1);
            q += __shfl_xor(q, 2);
            q += __shfl_xor(q, 4);
            q += __shfl_xor(q, 8);
            tsum[r] = q;
        }
        if (l16 == 0) {
#pragma unroll
            for (int r = 0; r < 8; ++r) sSum[wid][r + half * 8] = tsum[r];
        }
        __syncthreads();

#pragma unroll
        for (int r = 0; r < 8; ++r) {
            int row = r + half * 8;
            float ladd = sSum[0][row] + sSum[1][row] + sSum[2][row] + sSum[3][row];
            lrow[r] = lrow[r] * scl[r] + ladd;
            mrow[r] = mnew[r];
#pragma unroll
            for (int t = 0; t < 4; ++t) acc[t][r] *= scl[r];
        }

        // ---- out(q x c) += P(q x j) * v(j x c), K = 64 in two chunks ----
#pragma unroll
        for (int jj = 0; jj < 2; ++jj) {
            bf16x16 ap;
            {
                const __bf16* prow = &sP[l16][jj * 32 + half * 16];
#pragma unroll
                for (int e = 0; e < 16; ++e) ap[e] = prow[e];
            }
#pragma unroll
            for (int t = 0; t < 4; ++t) {
                int c = wid * 64 + t * 16 + l16;
                bf16x16 bv = frag_b_lds(&sV[buf][c * 64 + jj * 32 + half * 16]);
                acc[t] = WMMA_BF16(ap, bv, acc[t]);
            }
        }
        __syncthreads();                             // sP / LDS bufs reused
    }

    // ---- normalize by row sum, store aT (B, N, C) for the proj GEMM ----
#pragma unroll
    for (int r = 0; r < 8; ++r) {
        float inv = 1.f / lrow[r];
        int q = i0 + r + half * 8;
#pragma unroll
        for (int t = 0; t < 4; ++t) {
            int c = wid * 64 + t * 16 + l16;
            aT[((size_t)b * NSEQ + q) * CCH + c] = (__bf16)(acc[t][r] * inv);
        }
    }
}

// ---------------------------------------------------------------------------
// Kernel 4: output projection + bias + residual -> fp32 d_out (B, C, N).
// One wave per 32x32 tile (2x2 WMMA tiles).
// ---------------------------------------------------------------------------
__global__ __launch_bounds__(128)
void proj_kernel(const __bf16* __restrict__ aT, const __bf16* __restrict__ W,
                 const float* __restrict__ bias, const float* __restrict__ x,
                 float* __restrict__ out) {
    const int wid  = threadIdx.x >> 5;
    const int lane = threadIdx.x & 31;
    const int half = lane >> 4;
    const int l16  = lane & 15;

    const int tile = blockIdx.x * 4 + wid;           // 32 * 8 * 32 waves
    const int b    = tile / (8 * 32);
    const int rem  = tile % (8 * 32);
    const int ob   = rem / 32;
    const int nb   = rem % 32;

    const __bf16* arow0 = W + (size_t)(ob * 32 + l16) * CCH;
    const __bf16* arow1 = arow0 + (size_t)16 * CCH;
    const __bf16* bcol0 = aT + ((size_t)b * NSEQ + nb * 32 + l16) * CCH;
    const __bf16* bcol1 = bcol0 + (size_t)16 * CCH;

    f32x8 d00 = {}, d01 = {}, d10 = {}, d11 = {};
#pragma unroll
    for (int kk = 0; kk < 8; ++kk) {
        bf16x16 a0 = frag_a_bf16(arow0, kk, half);
        bf16x16 a1 = frag_a_bf16(arow1, kk, half);
        bf16x16 b0 = frag_b_bf16(bcol0, kk, half);
        bf16x16 b1 = frag_b_bf16(bcol1, kk, half);
        d00 = WMMA_BF16(a0, b0, d00);
        d01 = WMMA_BF16(a0, b1, d01);
        d10 = WMMA_BF16(a1, b0, d10);
        d11 = WMMA_BF16(a1, b1, d11);
    }

#pragma unroll
    for (int ti = 0; ti < 2; ++ti) {
#pragma unroll
        for (int tj = 0; tj < 2; ++tj) {
            const f32x8& d = ti ? (tj ? d11 : d10) : (tj ? d01 : d00);
            int n = nb * 32 + tj * 16 + l16;
#pragma unroll
            for (int r = 0; r < 8; ++r) {
                int o = ob * 32 + ti * 16 + r + half * 8;
                size_t idx = ((size_t)b * CCH + o) * NSEQ + n;
                out[idx] = d[r] + bias[o] + x[idx];
            }
        }
    }
}

// ---------------------------------------------------------------------------
extern "C" void kernel_launch(void* const* d_in, const int* in_sizes, int n_in,
                              void* d_out, int out_size, void* d_ws, size_t ws_size,
                              hipStream_t stream) {
    const float* x      = (const float*)d_in[0];
    const float* gn_w   = (const float*)d_in[1];
    const float* gn_b   = (const float*)d_in[2];
    const float* qkv_w  = (const float*)d_in[3];
    const float* qkv_b  = (const float*)d_in[4];
    const float* proj_w = (const float*)d_in[5];
    const float* proj_b = (const float*)d_in[6];
    float* out = (float*)d_out;

    // Workspace layout (bf16): xnT/aT share a buffer (xnT dead after QKV).
    const size_t ELEMS = (size_t)BATCH * CCH * NSEQ;            // 8 M
    __bf16* xnT = (__bf16*)d_ws;                                 // 16 MB
    __bf16* qT  = xnT + ELEMS;                                   // 16 MB
    __bf16* kT  = qT + ELEMS;                                    // 16 MB
    __bf16* v   = kT + ELEMS;                                    // 16 MB
    __bf16* wq  = v + ELEMS;                                     // 384 KB
    __bf16* wp  = wq + (size_t)O3 * CCH;                         // 128 KB
    __bf16* aT  = xnT;                                           // reuse

    wcvt_kernel<<<(O3 * CCH) / 256, 256, 0, stream>>>(qkv_w, proj_w, wq, wp);

    gn_kernel<<<BATCH * GRP, 256, 0, stream>>>(x, gn_w, gn_b, xnT);

    // 32 * 24 * 32 = 24576 waves, 4 waves/block
    qkv_kernel<<<(BATCH * 24 * 32) / 4, 128, 0, stream>>>(xnT, wq, qkv_b,
                                                          qT, kT, v);

    flash_kernel<<<BATCH * (NSEQ / 16), 128, 0, stream>>>(qT, kT, v, aT);

    // 32 * 8 * 32 = 8192 waves, 4 waves/block
    proj_kernel<<<(BATCH * 8 * 32) / 4, 128, 0, stream>>>(aT, wp, proj_b, x, out);
}
